// MS_MLP_Conv_18305150615567
// MI455X (gfx1250) — matrix operand order repaired and checked by reading, not compiled, not run on hardware
//
#include <hip/hip_runtime.h>
#include <hip/hip_bf16.h>

typedef __attribute__((ext_vector_type(16))) _Float16 v16h;
typedef __attribute__((ext_vector_type(8)))  _Float16 v8h;
typedef __attribute__((ext_vector_type(8)))  float    v8f;

#define T_    4
#define B_    64
#define C_    256
#define HID_  1024
#define HW_   196
#define NPIX_ (T_*B_*HW_)              /* 50176 */
#define TSTRIDE_ ((size_t)B_*C_*HW_)   /* 3211264: per-t element stride in x and S1 */
#define HSTRIDE_ ((size_t)B_*HW_*HID_) /* 12845056: per-t element stride in H1/S2 */

__device__ __forceinline__ v16h concat16(v8h lo, v8h hi) {
  return __builtin_shufflevector(lo, hi, 0,1,2,3,4,5,6,7,8,9,10,11,12,13,14,15);
}

// ---------------- utility kernels ----------------
__global__ void zero_stats_k(float* stats, int n) {
  for (int i = threadIdx.x; i < n; i += blockDim.x) stats[i] = 0.f;
}

__global__ void cvt_w_k(const float* __restrict__ w1, const float* __restrict__ w2,
                        _Float16* __restrict__ wt, int n1, int n2) {
  int i = blockIdx.x * blockDim.x + threadIdx.x;
  if (i < n1) wt[i] = (_Float16)w1[i];
  else if (i < n1 + n2) wt[i] = (_Float16)w2[i - n1];
}

// ---------------- LIF over x, transpose [T,B,C,HW] -> pixel-major f16 [NPIX, C] ----------------
__global__ void lif1_k(const float* __restrict__ x, _Float16* __restrict__ S1) {
  __shared__ _Float16 tile[32][66];
  const int hw0 = blockIdx.x * 64;
  const int c0  = blockIdx.y * 32;
  const int b   = blockIdx.z;
  const int tid = threadIdx.x;
  const int hw_l = tid & 63;
  const int c_r  = tid >> 6;                 // 0..3
  const int hw   = hw0 + hw_l;
  const bool hwok = hw < HW_;
  float v[8];
  #pragma unroll
  for (int k = 0; k < 8; ++k) v[k] = 0.f;

  for (int t = 0; t < T_; ++t) {
    #pragma unroll
    for (int k = 0; k < 8; ++k) {
      int cl = c_r * 8 + k;                  // 0..31
      float s = 0.f;
      if (hwok) {
        float xv = x[(size_t)t * TSTRIDE_ + ((size_t)(b * C_ + c0 + cl)) * HW_ + hw];
        float vv = 0.5f * (v[k] + xv);       // v += (x - v)/tau, tau=2
        s    = (vv >= 1.f) ? 1.f : 0.f;
        v[k] = (vv >= 1.f) ? 0.f : vv;       // hard reset
      }
      tile[cl][hw_l] = (_Float16)s;
    }
    __syncthreads();
    const int c_w  = tid & 31;
    const int hw_r = tid >> 5;               // 0..7
    #pragma unroll
    for (int p = 0; p < 8; ++p) {
      int hwl = p * 8 + hw_r;
      int hww = hw0 + hwl;
      if (hww < HW_)
        S1[(size_t)t * TSTRIDE_ + ((size_t)(b * HW_ + hww)) * C_ + c0 + c_w] = tile[c_w][hwl];
    }
    __syncthreads();
  }
}

// ---------------- WMMA GEMM: C[M,N] = A[M,K](f16,row-major) * B[N,K]^T + bias ----------------
// per-wave 16(M) x 64(N) tile; per-lane A(16x32)/B(32x16) loads use contiguous k = 16*hi + s for both (not AMD's table
// order). Register double-buffered K-loop so next-step loads overlap current WMMAs.
// Consecutive waves take consecutive M tiles with the SAME N tile -> B cacheline reuse in WGP$.
__global__ void gemm_k(const _Float16* __restrict__ A, const _Float16* __restrict__ Bw,
                       const float* __restrict__ bias, float* __restrict__ C,
                       int M, int N, int K) {
  const int lane = threadIdx.x & 31;
  const int hi   = lane >> 4;                // half-wave select
  const int nl   = lane & 15;
  const int wave = (blockIdx.x * blockDim.x + threadIdx.x) >> 5;
  const int mtiles = M >> 4;
  if (wave >= mtiles * (N >> 6)) return;     // wave-uniform
  const int m0 = (wave % mtiles) << 4;       // consecutive waves -> consecutive M
  const int n0 = (wave / mtiles) << 6;       // ... sharing one N tile (B reuse)

  // A lane row: lanes 0-15 hold M=lane (K=0..15); lanes 16-31 hold M=lane-16 (K=16..31)
  const _Float16* arow = A + (size_t)(m0 + nl) * K + hi * 16;
  // B lane: lanes 0-15 hold N=lane, K=k0..k0+15; lanes 16-31 hold N=lane-16, K=k0+16..k0+31
  const _Float16* brow0 = Bw + (size_t)(n0 + 0 * 16 + nl) * K + hi * 16;
  const _Float16* brow1 = Bw + (size_t)(n0 + 1 * 16 + nl) * K + hi * 16;
  const _Float16* brow2 = Bw + (size_t)(n0 + 2 * 16 + nl) * K + hi * 16;
  const _Float16* brow3 = Bw + (size_t)(n0 + 3 * 16 + nl) * K + hi * 16;

  v8f acc[4] = {};

  // prologue: load k-step 0
  v16h a_cur = concat16(*(const v8h*)(arow), *(const v8h*)(arow + 8));
  v16h b_cur[4];
  b_cur[0] = concat16(*(const v8h*)(brow0), *(const v8h*)(brow0 + 8));
  b_cur[1] = concat16(*(const v8h*)(brow1), *(const v8h*)(brow1 + 8));
  b_cur[2] = concat16(*(const v8h*)(brow2), *(const v8h*)(brow2 + 8));
  b_cur[3] = concat16(*(const v8h*)(brow3), *(const v8h*)(brow3 + 8));

  for (int k0 = 32; k0 < K; k0 += 32) {
    // issue next-step loads before consuming current registers
    v16h a_nxt = concat16(*(const v8h*)(arow + k0), *(const v8h*)(arow + k0 + 8));
    v16h b_nxt[4];
    b_nxt[0] = concat16(*(const v8h*)(brow0 + k0), *(const v8h*)(brow0 + k0 + 8));
    b_nxt[1] = concat16(*(const v8h*)(brow1 + k0), *(const v8h*)(brow1 + k0 + 8));
    b_nxt[2] = concat16(*(const v8h*)(brow2 + k0), *(const v8h*)(brow2 + k0 + 8));
    b_nxt[3] = concat16(*(const v8h*)(brow3 + k0), *(const v8h*)(brow3 + k0 + 8));
    #pragma unroll
    for (int j = 0; j < 4; ++j)
      acc[j] = __builtin_amdgcn_wmma_f32_16x16x32_f16(
          false, a_cur, false, b_cur[j], (short)0, acc[j], false, false);
    a_cur = a_nxt;
    #pragma unroll
    for (int j = 0; j < 4; ++j) b_cur[j] = b_nxt[j];
  }
  // epilogue k-step
  #pragma unroll
  for (int j = 0; j < 4; ++j)
    acc[j] = __builtin_amdgcn_wmma_f32_16x16x32_f16(
        false, a_cur, false, b_cur[j], (short)0, acc[j], false, false);

  // C/D layout: VGPR r -> M = r + 8*hi, N = n0 + j*16 + (lane&15)
  #pragma unroll
  for (int j = 0; j < 4; ++j) {
    int n = n0 + j * 16 + nl;
    float bb = bias[n];
    float* cp = C + (size_t)(m0 + hi * 8) * N + n;
    #pragma unroll
    for (int r = 0; r < 8; ++r) cp[(size_t)r * N] = acc[j][r] + bb;
  }
}

// ---------------- per-channel sum / sumsq over pixels ----------------
__global__ void reduce_k(const float* __restrict__ H, float* __restrict__ sum,
                         float* __restrict__ sq, int Nc) {
  int c = blockIdx.x * blockDim.x + threadIdx.x;       // gridDim.x = Nc/256
  size_t p0 = (size_t)blockIdx.y * 512;                // gridDim.y = 98; 98*512 = 50176
  float s = 0.f, q = 0.f;
  for (int p = 0; p < 512; ++p) {
    float v = H[(p0 + p) * Nc + c];
    s += v; q += v * v;
  }
  atomicAdd(&sum[c], s);
  atomicAdd(&sq[c],  q);
}

__global__ void finalize_k(float* sum, float* sq, const float* __restrict__ g,
                           const float* __restrict__ be, int Nc, float invNp) {
  int c = blockIdx.x * blockDim.x + threadIdx.x;
  if (c < Nc) {
    float mean = sum[c] * invNp;
    float var  = sq[c] * invNp - mean * mean;
    float sc   = g[c] * rsqrtf(var + 1e-5f);
    float sh   = be[c] - mean * sc;
    sum[c] = sc;   // fold BN into scale/shift, in place
    sq[c]  = sh;
  }
}

// ---------------- fused BN + LIF on hidden, f32 -> f16 spikes ----------------
__global__ void bnlif2_k(const float* __restrict__ H1, const float* __restrict__ sc,
                         const float* __restrict__ sh, _Float16* __restrict__ S2) {
  size_t i = (size_t)blockIdx.x * blockDim.x + threadIdx.x;
  if (i >= HSTRIDE_) return;
  int n = (int)(i & (HID_ - 1));
  float a = sc[n], b = sh[n];
  float v = 0.f;
  #pragma unroll
  for (int t = 0; t < T_; ++t) {
    float h  = H1[i + (size_t)t * HSTRIDE_] * a + b;
    float vv = 0.5f * (v + h);
    float s  = (vv >= 1.f) ? 1.f : 0.f;
    v        = (vv >= 1.f) ? 0.f : vv;
    S2[i + (size_t)t * HSTRIDE_] = (_Float16)s;
  }
}

// ---------------- final: BN2 + residual, transpose [NPIX,C] -> [T,B,C,HW] ----------------
__global__ void out_k(const float* __restrict__ H2, const float* __restrict__ sc2,
                      const float* __restrict__ sh2, const float* __restrict__ x,
                      float* __restrict__ out) {
  __shared__ float tile[64][65];
  const int tb  = blockIdx.y;          // t*B + b
  const int hw0 = (blockIdx.x & 3) * 64;
  const int c0  = (blockIdx.x >> 2) * 64;
  const int tid = threadIdx.x;
  {
    int c_l = tid & 63, hw_r = tid >> 6;
    float a = sc2[c0 + c_l], b = sh2[c0 + c_l];
    #pragma unroll
    for (int p = 0; p < 16; ++p) {
      int hwl = p * 4 + hw_r;
      int hw  = hw0 + hwl;
      if (hw < HW_) {
        float v = H2[((size_t)tb * HW_ + hw) * C_ + c0 + c_l];
        tile[c_l][hwl] = v * a + b;
      }
    }
  }
  __syncthreads();
  {
    int hw_l = tid & 63, c_r = tid >> 6;
    int hw = hw0 + hw_l;
    #pragma unroll
    for (int p = 0; p < 16; ++p) {
      int cl = p * 4 + c_r;
      if (hw < HW_) {
        size_t oi = ((size_t)tb * C_ + c0 + cl) * HW_ + hw;
        out[oi] = tile[cl][hw_l] + x[oi];
      }
    }
  }
}

// ---------------- launch ----------------
extern "C" void kernel_launch(void* const* d_in, const int* in_sizes, int n_in,
                              void* d_out, int out_size, void* d_ws, size_t ws_size,
                              hipStream_t stream) {
  const float* x   = (const float*)d_in[0];
  const float* w1  = (const float*)d_in[1];
  const float* b1  = (const float*)d_in[2];
  const float* g1  = (const float*)d_in[3];
  const float* be1 = (const float*)d_in[4];
  const float* w2  = (const float*)d_in[5];
  const float* b2  = (const float*)d_in[6];
  const float* g2  = (const float*)d_in[7];
  const float* be2 = (const float*)d_in[8];
  float* out = (float*)d_out;

  char* ws = (char*)d_ws;
  const size_t OFF_WT1   = 0;                                   // 1024*256 f16
  const size_t OFF_WT2   = OFF_WT1 + (size_t)HID_*C_*2;         // 256*1024 f16
  const size_t OFF_S1    = OFF_WT2 + (size_t)C_*HID_*2;         // NPIX*C f16
  const size_t OFF_S2    = OFF_S1  + (size_t)NPIX_*C_*2;        // NPIX*HID f16
  const size_t OFF_H1    = OFF_S2  + (size_t)NPIX_*HID_*2;      // NPIX*HID f32
  const size_t OFF_STATS = OFF_H1  + (size_t)NPIX_*HID_*4;      // 2560 f32
  // H2 aliases H1 region (H1 dead after bnlif2)
  _Float16* Wt1 = (_Float16*)(ws + OFF_WT1);
  _Float16* Wt2 = (_Float16*)(ws + OFF_WT2);
  _Float16* S1  = (_Float16*)(ws + OFF_S1);
  _Float16* S2  = (_Float16*)(ws + OFF_S2);
  float*    H1  = (float*)(ws + OFF_H1);
  float*    H2  = (float*)(ws + OFF_H1);
  float*    st  = (float*)(ws + OFF_STATS);
  float *sum1 = st, *sq1 = st + 1024, *sum2 = st + 2048, *sq2 = st + 2304;

  const float invNp = 1.f / (float)NPIX_;

  zero_stats_k<<<1, 256, 0, stream>>>(st, 2560);
  cvt_w_k<<<(2 * HID_ * C_ + 255) / 256, 256, 0, stream>>>(w1, w2, Wt1, HID_ * C_, C_ * HID_);

  lif1_k<<<dim3(4, C_ / 32, B_), 256, 0, stream>>>(x, S1);

  // GEMM1: [50176 x 256] * [256 x 1024]; 16x64 tile/wave -> 50176 waves / 8 per block
  gemm_k<<<(NPIX_ / 16) * (HID_ / 64) / 8, 256, 0, stream>>>(S1, Wt1, b1, H1, NPIX_, HID_, C_);

  reduce_k<<<dim3(HID_ / 256, 98), 256, 0, stream>>>(H1, sum1, sq1, HID_);
  finalize_k<<<HID_ / 256, 256, 0, stream>>>(sum1, sq1, g1, be1, HID_, invNp);

  bnlif2_k<<<(unsigned)(HSTRIDE_ / 256), 256, 0, stream>>>(H1, sum1, sq1, S2);

  // GEMM2: [50176 x 1024] * [1024 x 256]
  gemm_k<<<(NPIX_ / 16) * (C_ / 64) / 8, 256, 0, stream>>>(S2, Wt2, b2, H2, NPIX_, C_, HID_);

  reduce_k<<<dim3(C_ / 256, 98), 256, 0, stream>>>(H2, sum2, sq2, C_);
  finalize_k<<<1, 256, 0, stream>>>(sum2, sq2, g2, be2, C_, invNp);

  out_k<<<dim3(16, T_ * B_), 256, 0, stream>>>(H2, sum2, sq2, x, out);
}
